// SelfAttention_88399016886917
// MI455X (gfx1250) — matrix operand order repaired and checked by reading, not compiled, not run on hardware
//
#include <hip/hip_runtime.h>
#include <hip/hip_bf16.h>

// MI455X (gfx1250) fused multi-head self-attention.
// Pipeline: bitmask pack (ballot, 32x less mask HBM traffic) ->
// per-head QKV projection (f16 WMMA, f32 acc) -> flash-attention with 32-row
// q tiles (K/V fragments reused across two M-tiles; online softmax in exp2
// domain; LDS bounce for D->A layout) -> output GEMM with 32-row tiles.
// Workspace: Qp/Kp f16 [N,H,L,Dh], Vt f16 [N,H,Dh,L], AO f16 [N,L,D], mask
// bits [N,L,L/32] = 3*8MB + 8MB + 1MB ~ 34.5 MB, fully L2 resident (192 MB).

typedef __attribute__((ext_vector_type(16))) _Float16 v16h;
typedef __attribute__((ext_vector_type(8)))  _Float16 v8h;
typedef __attribute__((ext_vector_type(8)))  float    v8f;
typedef __attribute__((ext_vector_type(4)))  float    v4f;

#define L_SEQ   2048
#define D_HEAD  64
#define N_HEADS 16
#define N_BATCH 2
#define D_EMB   1024
#define LW      (L_SEQ / 32)                 // mask words per row

static __device__ __forceinline__ v8f wmma_f16(v16h a, v16h b, v8f c) {
  return __builtin_amdgcn_wmma_f32_16x16x32_f16(false, a, false, b, (short)0, c,
                                                false, false);
}

// ---- A fragment (16x32 f16): lane holds row (lane&15); element e has
// K = (e/8)*16 + ((lane>>4)*8) + (e&7). Two contiguous 8-half runs per lane.
static __device__ __forceinline__ v16h load_a_f16(const _Float16* base, int ld) {
  int lane = threadIdx.x & 31;
  int row  = lane & 15;
  int hk   = (lane >> 4) << 3;
  const _Float16* r = base + (size_t)row * ld + hk;
  v8h lo = *(const v8h*)(r);
  v8h hi = *(const v8h*)(r + 16);
  v16h a;
#pragma unroll
  for (int i = 0; i < 8; i++) { a[i] = lo[i]; a[i + 8] = hi[i]; }
  return a;
}

static __device__ __forceinline__ v16h load_a_f32(const float* base, int ld) {
  int lane = threadIdx.x & 31;
  int row  = lane & 15;
  int hk   = (lane >> 4) << 3;
  const float* r = base + (size_t)row * ld + hk;
  v4f x0 = *(const v4f*)(r);
  v4f x1 = *(const v4f*)(r + 4);
  v4f x2 = *(const v4f*)(r + 16);
  v4f x3 = *(const v4f*)(r + 20);
  v16h a;
#pragma unroll
  for (int i = 0; i < 4; i++) {
    a[i]      = (_Float16)x0[i];
    a[4 + i]  = (_Float16)x1[i];
    a[8 + i]  = (_Float16)x2[i];
    a[12 + i] = (_Float16)x3[i];
  }
  return a;
}

// ---- B fragment (32x16): lane holds col (lane&15); K = (e/8)*16 + ((lane>>4)*8) + (e&7).
// Memory holds B transposed: element (k,n) at base[n*ld + k].
static __device__ __forceinline__ v16h load_b_f16_t(const _Float16* base, int ld) {
  int lane = threadIdx.x & 31;
  int col  = lane & 15;
  int kk   = (lane >> 4) << 3;
  const _Float16* r = base + (size_t)col * ld + kk;
  v8h lo = *(const v8h*)(r);
  v8h hi = *(const v8h*)(r + 16);
  v16h b;
#pragma unroll
  for (int i = 0; i < 8; i++) { b[i] = lo[i]; b[i + 8] = hi[i]; }
  return b;
}

static __device__ __forceinline__ v16h load_b_f32_t(const float* base, int ld) {
  int lane = threadIdx.x & 31;
  int col  = lane & 15;
  int kk   = (lane >> 4) << 3;
  const float* r = base + (size_t)col * ld + kk;
  v4f x0 = *(const v4f*)(r);
  v4f x1 = *(const v4f*)(r + 4);
  v4f x2 = *(const v4f*)(r + 16);
  v4f x3 = *(const v4f*)(r + 20);
  v16h b;
#pragma unroll
  for (int i = 0; i < 4; i++) {
    b[i]      = (_Float16)x0[i];
    b[4 + i]  = (_Float16)x1[i];
    b[8 + i]  = (_Float16)x2[i];
    b[12 + i] = (_Float16)x3[i];
  }
  return b;
}

// ============ Kernel 0: pack mask int32 -> 1 bit/elem via wave32 ballot ======
__global__ void maskpack_kernel(const int* __restrict__ mask,
                                unsigned* __restrict__ bits) {
  size_t i = (size_t)blockIdx.x * blockDim.x + threadIdx.x;
  unsigned b = (unsigned)__ballot(mask[i] != 0);
  if ((threadIdx.x & 31) == 0) bits[i >> 5] = b;
}

// ============ Kernel 1: per-head linear projection  out = x @ W^T ============
__global__ void proj_kernel(const float* __restrict__ src,
                            const float* __restrict__ W,
                            _Float16* __restrict__ dst, int transposeOut) {
  int wave = (int)((blockIdx.x * blockDim.x + threadIdx.x) >> 5);
  int lane = threadIdx.x & 31;
  int rt = wave & 127;
  int h  = (wave >> 7) & 15;
  int n  = wave >> 11;
  int l0 = rt * 16;

  const float* sp = src + ((size_t)n * L_SEQ + l0) * D_EMB + h * D_HEAD;
  v16h a0 = load_a_f32(sp, D_EMB);
  v16h a1 = load_a_f32(sp + 32, D_EMB);

  int cl = lane & 15;
  int hi = (lane >> 4) << 3;
#pragma unroll
  for (int nt = 0; nt < 4; nt++) {
    v16h b0 = load_b_f32_t(W + (size_t)nt * 16 * D_HEAD, D_HEAD);
    v16h b1 = load_b_f32_t(W + (size_t)nt * 16 * D_HEAD + 32, D_HEAD);
    v8f c = {};
    c = wmma_f16(a0, b0, c);
    c = wmma_f16(a1, b1, c);
    if (!transposeOut) {                     // [N,H,L,Dh]
      _Float16* o = dst + (((size_t)n * N_HEADS + h) * L_SEQ + l0) * D_HEAD +
                    nt * 16 + cl;
#pragma unroll
      for (int j = 0; j < 8; j++) o[(size_t)(j + hi) * D_HEAD] = (_Float16)c[j];
    } else {                                 // V transposed: [N,H,Dh,L]
      _Float16* o = dst + (((size_t)n * N_HEADS + h) * D_HEAD + nt * 16 + cl) *
                    L_SEQ + l0 + hi;
#pragma unroll
      for (int j = 0; j < 8; j++) o[j] = (_Float16)c[j];
    }
  }
}

// ==== Kernel 2: flash attention, one wave per (n, h, 32-row q tile) ==========
__global__ void attn_kernel(const _Float16* __restrict__ Qp,
                            const _Float16* __restrict__ Kp,
                            const _Float16* __restrict__ Vt,
                            const unsigned* __restrict__ mbits,
                            _Float16* __restrict__ AO) {
  __shared__ _Float16 ldsP[4 * 2 * 16 * 32]; // 4 waves x 2 M-tiles x 16x32
  int wv   = threadIdx.x >> 5;
  int lane = threadIdx.x & 31;
  int wave = (int)(blockIdx.x * (blockDim.x >> 5)) + wv;
  int rt = wave & 63;                        // L/32 = 64 q tiles
  int h  = (wave >> 6) & 15;
  int n  = wave >> 10;
  int q0 = rt * 32;

  const _Float16* Qh = Qp + (((size_t)n * N_HEADS + h) * L_SEQ + q0) * D_HEAD;
  const _Float16* Kh = Kp + ((size_t)n * N_HEADS + h) * L_SEQ * D_HEAD;
  const _Float16* Vh = Vt + ((size_t)n * N_HEADS + h) * D_HEAD * L_SEQ;
  const unsigned* mrow = mbits + (size_t)(n * L_SEQ + q0) * LW;

  v16h qa[2][2];
#pragma unroll
  for (int mt = 0; mt < 2; mt++) {
    qa[mt][0] = load_a_f16(Qh + (size_t)mt * 16 * D_HEAD, D_HEAD);
    qa[mt][1] = load_a_f16(Qh + (size_t)mt * 16 * D_HEAD + 32, D_HEAD);
  }

  int cl = lane & 15;
  int hi = (lane >> 4) << 3;
  unsigned bit0 = 1u << cl;
  unsigned bit1 = bit0 << 16;
  float m[2][8], l[2][8];
  v8f acc[2][4];
#pragma unroll
  for (int mt = 0; mt < 2; mt++) {
#pragma unroll
    for (int j = 0; j < 8; j++) { m[mt][j] = -3.0e38f; l[mt][j] = 0.0f; }
#pragma unroll
    for (int t = 0; t < 4; t++) {
      v8f z = {};
      acc[mt][t] = z;
    }
  }
  _Float16* P = ldsP + wv * (2 * 16 * 32);
  // energy/sqrt(D_EMB) in exp2 domain: scale = log2(e)/32
  const float sc2 = 1.4426950408889634f * 0.03125f;

  for (int kv = 0; kv < L_SEQ; kv += 32) {
    // K fragments shared by both M-tiles
    v16h kb0 = load_b_f16_t(Kh + (size_t)kv * D_HEAD, D_HEAD);
    v16h kb1 = load_b_f16_t(Kh + (size_t)kv * D_HEAD + 32, D_HEAD);
    v16h kb2 = load_b_f16_t(Kh + (size_t)(kv + 16) * D_HEAD, D_HEAD);
    v16h kb3 = load_b_f16_t(Kh + (size_t)(kv + 16) * D_HEAD + 32, D_HEAD);
#pragma unroll
    for (int mt = 0; mt < 2; mt++) {
      v8f s0 = {}, s1 = {};
      s0 = wmma_f16(qa[mt][0], kb0, s0);
      s0 = wmma_f16(qa[mt][1], kb1, s0);
      s1 = wmma_f16(qa[mt][0], kb2, s1);
      s1 = wmma_f16(qa[mt][1], kb3, s1);
      float p0[8], p1[8];
#pragma unroll
      for (int j = 0; j < 8; j++) {
        unsigned mw = mrow[(size_t)(mt * 16 + j + hi) * LW + (kv >> 5)];
        float x0 = (mw & bit0) ? s0[j] * sc2 : -1e20f;
        float x1 = (mw & bit1) ? s1[j] * sc2 : -1e20f;
        float tm = fmaxf(x0, x1);
#pragma unroll
        for (int off = 8; off; off >>= 1)
          tm = fmaxf(tm, __shfl_xor(tm, off, 32));
        float mn = fmaxf(m[mt][j], tm);
        float f  = exp2f(m[mt][j] - mn);
        p0[j] = exp2f(x0 - mn);
        p1[j] = exp2f(x1 - mn);
        l[mt][j] = l[mt][j] * f + p0[j] + p1[j];
        m[mt][j] = mn;
#pragma unroll
        for (int t = 0; t < 4; t++) acc[mt][t][j] *= f;
      }
#pragma unroll
      for (int j = 0; j < 8; j++) {
        P[mt * 512 + (j + hi) * 32 + cl]      = (_Float16)p0[j];
        P[mt * 512 + (j + hi) * 32 + cl + 16] = (_Float16)p1[j];
      }
    }
    asm volatile("s_wait_dscnt 0" ::: "memory");
    v16h pa[2];
#pragma unroll
    for (int mt = 0; mt < 2; mt++) {
      int row = lane & 15;
      int hk  = (lane >> 4) << 3;
      const _Float16* r = P + mt * 512 + row * 32 + hk;
      v8h lo = *(const v8h*)(r);
      v8h hh = *(const v8h*)(r + 16);
#pragma unroll
      for (int i = 0; i < 8; i++) { pa[mt][i] = lo[i]; pa[mt][i + 8] = hh[i]; }
    }
    asm volatile("" ::: "memory");
    // V fragments shared by both M-tiles (Vt transposed -> contiguous)
#pragma unroll
    for (int t = 0; t < 4; t++) {
      v16h vb = load_b_f16_t(Vh + (size_t)(t * 16) * L_SEQ + kv, L_SEQ);
      acc[0][t] = wmma_f16(pa[0], vb, acc[0][t]);
      acc[1][t] = wmma_f16(pa[1], vb, acc[1][t]);
    }
  }

#pragma unroll
  for (int mt = 0; mt < 2; mt++) {
#pragma unroll
    for (int j = 0; j < 8; j++) {
      float s = l[mt][j];
#pragma unroll
      for (int off = 8; off; off >>= 1) s += __shfl_xor(s, off, 32);
      l[mt][j] = 1.0f / s;
    }
#pragma unroll
    for (int t = 0; t < 4; t++) {
      _Float16* o = AO + ((size_t)n * L_SEQ + q0 + mt * 16) * D_EMB +
                    h * D_HEAD + t * 16 + cl;
#pragma unroll
      for (int j = 0; j < 8; j++)
        o[(size_t)(j + hi) * D_EMB] = (_Float16)(acc[mt][t][j] * l[mt][j]);
    }
  }
}

// ============ Kernel 3: output projection  out = AO @ Wo^T + bo ==============
__global__ void outproj_kernel(const _Float16* __restrict__ AO,
                               const float* __restrict__ Wo,
                               const float* __restrict__ bo,
                               float* __restrict__ out) {
  int wave = (int)((blockIdx.x * blockDim.x + threadIdx.x) >> 5);
  int lane = threadIdx.x & 31;
  int cg = wave & 15;                        // 16 column groups of 64
  int rtile = wave >> 4;                     // 128 row tiles of 32
  int r0 = rtile * 32, c0 = cg * 64;

  v8f acc[2][4];
#pragma unroll
  for (int mt = 0; mt < 2; mt++)
#pragma unroll
    for (int t = 0; t < 4; t++) {
      v8f z = {};
      acc[mt][t] = z;
    }
  for (int kc = 0; kc < D_EMB; kc += 32) {
    v16h a0 = load_a_f16(AO + (size_t)r0 * D_EMB + kc, D_EMB);
    v16h a1 = load_a_f16(AO + (size_t)(r0 + 16) * D_EMB + kc, D_EMB);
#pragma unroll
    for (int t = 0; t < 4; t++) {
      v16h b = load_b_f32_t(Wo + (size_t)(c0 + t * 16) * D_EMB + kc, D_EMB);
      acc[0][t] = wmma_f16(a0, b, acc[0][t]);
      acc[1][t] = wmma_f16(a1, b, acc[1][t]);
    }
  }
  int cl = lane & 15;
  int hi = (lane >> 4) << 3;
#pragma unroll
  for (int mt = 0; mt < 2; mt++)
#pragma unroll
    for (int t = 0; t < 4; t++) {
      int e = c0 + t * 16 + cl;
      float bias = bo[e];
      float* o = out + (size_t)(r0 + mt * 16) * D_EMB + e;
#pragma unroll
      for (int j = 0; j < 8; j++)
        o[(size_t)(j + hi) * D_EMB] = acc[mt][t][j] + bias;
    }
}

// ============================== launch =======================================
extern "C" void kernel_launch(void* const* d_in, const int* in_sizes, int n_in,
                              void* d_out, int out_size, void* d_ws,
                              size_t ws_size, hipStream_t stream) {
  const float* values = (const float*)d_in[0];
  const float* keys   = (const float*)d_in[1];
  const float* query  = (const float*)d_in[2];
  const int*   mask   = (const int*)d_in[3];
  const float* Wv     = (const float*)d_in[4];
  const float* Wk     = (const float*)d_in[5];
  const float* Wq     = (const float*)d_in[6];
  const float* Wo     = (const float*)d_in[7];
  const float* bo     = (const float*)d_in[8];
  float* out = (float*)d_out;

  char* ws = (char*)d_ws;
  const size_t SZ = (size_t)N_BATCH * N_HEADS * L_SEQ * D_HEAD * sizeof(_Float16);
  _Float16* Qp = (_Float16*)(ws);
  _Float16* Kp = (_Float16*)(ws + SZ);
  _Float16* Vt = (_Float16*)(ws + 2 * SZ);
  _Float16* AO = (_Float16*)(ws + 3 * SZ);
  unsigned* MB = (unsigned*)(ws + 4 * SZ);   // N*L*L/32 words = 1 MB

  dim3 blk(128);                             // 4 waves per block (wave32)
  size_t nmask = (size_t)N_BATCH * L_SEQ * L_SEQ;          // 8388608
  maskpack_kernel<<<dim3((unsigned)(nmask / 256)), dim3(256), 0, stream>>>(mask, MB);

  int pwaves = N_BATCH * N_HEADS * (L_SEQ / 16);           // 4096
  proj_kernel<<<dim3(pwaves / 4), blk, 0, stream>>>(query,  Wq, Qp, 0);
  proj_kernel<<<dim3(pwaves / 4), blk, 0, stream>>>(keys,   Wk, Kp, 0);
  proj_kernel<<<dim3(pwaves / 4), blk, 0, stream>>>(values, Wv, Vt, 1);

  int awaves = N_BATCH * N_HEADS * (L_SEQ / 32);           // 2048
  attn_kernel<<<dim3(awaves / 4), blk, 0, stream>>>(Qp, Kp, Vt, MB, AO);

  int gwaves = (N_BATCH * L_SEQ / 32) * (D_EMB / 64);      // 2048
  outproj_kernel<<<dim3(gwaves / 4), blk, 0, stream>>>(AO, Wo, bo, out);
}
